// LSTM_85005992722747
// MI455X (gfx1250) — matrix operand order repaired and checked by reading, not compiled, not run on hardware
//
#include <hip/hip_runtime.h>

typedef __attribute__((ext_vector_type(16))) __bf16 v16bf;
typedef __attribute__((ext_vector_type(8)))  __bf16 bf16x8;
typedef __attribute__((ext_vector_type(4)))  __bf16 bf16x4;
typedef __attribute__((ext_vector_type(8)))  float  v8f;
typedef __attribute__((ext_vector_type(4)))  float  f32x4;
typedef int gv4i __attribute__((vector_size(16)));   // matches builtin pointee type

#define T_STEPS 256
#define NB      128
#define HD      1024
#define NH      (NB * HD)      // 131072 elems per [N,H] state
#define G4      (4 * HD)       // 4096 gate rows
#define WELEM   (G4 * HD)      // 4,194,304 elems per layer weight
#define KBLK    256            // K columns staged per round (bf16)
#define MROWS   64             // rows staged per block

#if defined(__HIP_DEVICE_COMPILE__) && \
    __has_builtin(__builtin_amdgcn_global_load_async_to_lds_b128) && \
    __has_builtin(__builtin_amdgcn_s_wait_asynccnt)
#define USE_ASYNC_LDS 1
#endif

// ---------------- fast activations on CDNA5 trans units --------------------
__device__ __forceinline__ float fast_exp2(float x) {
#if defined(__HIP_DEVICE_COMPILE__) && __has_builtin(__builtin_amdgcn_exp2f)
    return __builtin_amdgcn_exp2f(x);        // raw v_exp_f32
#else
    return exp2f(x);
#endif
}
__device__ __forceinline__ float fast_sigmoid(float x) {
    return 1.0f / (1.0f + fast_exp2(-1.4426950408889634f * x));
}
__device__ __forceinline__ float fast_tanh(float x) {
#if defined(__HIP_DEVICE_COMPILE__) && __has_builtin(__builtin_amdgcn_tanhf)
    return __builtin_amdgcn_tanhf(x);        // v_tanh_f32
#else
    float e = fast_exp2(2.885390081777927f * x);   // exp2(2*log2e*x)
    return 1.0f - 2.0f / (e + 1.0f);               // branchless, saturates to +-1
#endif
}

// ---------------- fp32 -> bf16 conversion (one-time, parallel) -------------
__global__ void k_cvt_bf16(const float* __restrict__ s, __bf16* __restrict__ d, int n4) {
    int i  = blockIdx.x * blockDim.x + threadIdx.x;
    int st = gridDim.x * blockDim.x;
    for (; i < n4; i += st) {
        f32x4 v = ((const f32x4*)s)[i];
        bf16x4 b;
        b.x = (__bf16)v.x; b.y = (__bf16)v.y; b.z = (__bf16)v.z; b.w = (__bf16)v.w;
        ((bf16x4*)d)[i] = b;
    }
}

__global__ void k_bias(const float* __restrict__ a, const float* __restrict__ b,
                       float* __restrict__ o, int n) {
    int i = blockIdx.x * blockDim.x + threadIdx.x;
    if (i < n) o[i] = a[i] + b[i];
}

// ---------------- fused LSTM cell: gates GEMM + activations ----------------
// grid: 128 blocks x 128 threads (4 waves, 1 per SIMD32).
// block bx: hidden cols [(bx>>1)*16, +16), rows [(bx&1)*64, +64), all 4 gates.
// wave w: rows [(bx&1)*64 + w*16, +16).
// LDS: double-buffered 64 x 256 bf16 A tile (2 x 32 KB), XOR-swizzled 16B chunks:
//   chunk c of local row r lives at halves  r*256 + ((c ^ (r & 15)) * 8)
// Pipeline: async-stage round b+1 into buf^1, compute round b from buf,
//           then s_wait_asynccnt + barrier.
__launch_bounds__(128)
__global__ void k_lstm_cell(const __bf16* __restrict__ xinb,   // [128,1024] bf16
                            const __bf16* __restrict__ hprevb, // [128,1024] bf16
                            float*        __restrict__ cstate, // [128,1024] f32 in/out
                            const __bf16* __restrict__ wih,    // [4096,1024] bf16
                            const __bf16* __restrict__ whh,    // [4096,1024] bf16
                            const float*  __restrict__ bias,   // [4096] f32
                            __bf16*       __restrict__ houtb,  // [128,1024] bf16
                            float*        __restrict__ houtf)  // [128,1024] f32
{
    __shared__ __bf16 smA[2][MROWS * KBLK];   // 2 x 32768 bytes

    const int tid   = threadIdx.x;
    const int lane  = tid & 31;
    const int wv    = tid >> 5;               // 0..3
    const int mrow0 = (blockIdx.x & 1) * MROWS;   // global row base of block
    const int mb    = wv * 16;                // local row base of this wave
    const int hb    = (blockIdx.x >> 1) * 16; // hidden-column base
    const int lrow  = lane & 15;
    const int lhi   = lane >> 4;              // 0/1 lane half

    // round b covers fused K = [x(0..3) | h(4..7)] * 256
    auto stage = [&](int b, int buf) {
        const __bf16* asrc = (b < 4) ? xinb : hprevb;
        const int     k0   = (b & 3) * KBLK;
        #pragma unroll
        for (int it = 0; it < 16; ++it) {
            int idx = it * 128 + tid;         // 0..2047 chunk slots
            int row = idx >> 5;               // 0..63 local row
            int c8  = idx & 31;               // 16B chunk within row
            int sw  = (c8 ^ (row & 15)) * 8;
            const __bf16* gp = asrc + (size_t)(mrow0 + row) * HD + k0 + c8 * 8;
            __bf16*       lp = &smA[buf][row * KBLK + sw];
#ifdef USE_ASYNC_LDS
            __builtin_amdgcn_global_load_async_to_lds_b128(
                (__attribute__((address_space(1))) gv4i*)gp,
                (__attribute__((address_space(3))) gv4i*)lp, 0, 0);
#else
            *(bf16x8*)lp = *(const bf16x8*)gp;
#endif
        }
    };

    v8f acc0 = {}, acc1 = {}, acc2 = {}, acc3 = {};   // i, f, g, o tiles

    // prologue: stage round 0 into buf 0
    stage(0, 0);
#ifdef USE_ASYNC_LDS
    __builtin_amdgcn_s_wait_asynccnt(0);
#endif
    __syncthreads();

    #pragma unroll 1
    for (int b = 0; b < 8; ++b) {
        const int     buf  = b & 1;
        const __bf16* wsrc = (b < 4) ? wih : whh;
        const int     k0   = (b & 3) * KBLK;
        const __bf16* smb  = smA[buf];

        // kick next round's async stage into the other buffer (overlaps compute)
        if (b < 7) stage(b + 1, buf ^ 1);

        const size_t wr = (size_t)(hb + lrow);
        const int    arow = mb + lrow;        // local row for A fragment

        auto loadA = [&](int kc) -> v16bf {
            v16bf a;
            int c0 = kc * 4 + lhi * 2;        // 16B chunk index
            ((bf16x8*)&a)[0] = *(const bf16x8*)&smb[arow * KBLK + ((c0    ) ^ (arow & 15)) * 8];
            ((bf16x8*)&a)[1] = *(const bf16x8*)&smb[arow * KBLK + ((c0 + 1) ^ (arow & 15)) * 8];
            return a;
        };
        auto loadB = [&](int kc, int q) -> v16bf {
            int kk = k0 + kc * 32 + lhi * 16;
            return *(const v16bf*)(wsrc + (wr + (size_t)q * 1024u) * HD + kk);
        };

        // software pipeline: fragments for kc+1 in flight under kc's 4 WMMAs
        v16bf aC  = loadA(0);
        v16bf b0C = loadB(0, 0), b1C = loadB(0, 1), b2C = loadB(0, 2), b3C = loadB(0, 3);
        #pragma unroll
        for (int kc = 0; kc < 8; ++kc) {
            v16bf aN = aC, b0N = b0C, b1N = b1C, b2N = b2C, b3N = b3C;
            if (kc < 7) {
                aN  = loadA(kc + 1);
                b0N = loadB(kc + 1, 0); b1N = loadB(kc + 1, 1);
                b2N = loadB(kc + 1, 2); b3N = loadB(kc + 1, 3);
            }
            acc0 = __builtin_amdgcn_wmma_f32_16x16x32_bf16(false, aC, false, b0C, (short)0, acc0, false, false);
            acc1 = __builtin_amdgcn_wmma_f32_16x16x32_bf16(false, aC, false, b1C, (short)0, acc1, false, false);
            acc2 = __builtin_amdgcn_wmma_f32_16x16x32_bf16(false, aC, false, b2C, (short)0, acc2, false, false);
            acc3 = __builtin_amdgcn_wmma_f32_16x16x32_bf16(false, aC, false, b3C, (short)0, acc3, false, false);
            aC = aN; b0C = b0N; b1C = b1N; b2C = b2N; b3C = b3N;
        }

        if (b < 7) {
#ifdef USE_ASYNC_LDS
            __builtin_amdgcn_s_wait_asynccnt(0);
#endif
            __syncthreads();
        }
    }

    // ---- fused epilogue: bias + sigmoid/tanh + c/h update (branchless) ----
    const int   col = hb + lrow;
    const float bi  = bias[col];
    const float bff = bias[col + 1024];
    const float bg  = bias[col + 2048];
    const float bo  = bias[col + 3072];

    #pragma unroll
    for (int r = 0; r < 8; ++r) {
        // C/D layout: VGPR r, lanes 0-15 -> M=r, lanes 16-31 -> M=8+r
        int row = mrow0 + mb + r + lhi * 8;
        float si = fast_sigmoid(acc0[r] + bi);
        float sf = fast_sigmoid(acc1[r] + bff);
        float tg = fast_tanh   (acc2[r] + bg);
        float so = fast_sigmoid(acc3[r] + bo);
        size_t off = (size_t)row * HD + col;
        float cn = sf * cstate[off] + si * tg;
        float hn = so * fast_tanh(cn);
        cstate[off] = cn;
        houtf[off]  = hn;
        houtb[off]  = (__bf16)hn;
    }
}

// ---------------------------------------------------------------------------
extern "C" void kernel_launch(void* const* d_in, const int* in_sizes, int n_in,
                              void* d_out, int out_size, void* d_ws, size_t ws_size,
                              hipStream_t stream) {
    const float* x   = (const float*)d_in[0];   // [256,128,1024]
    const float* h0  = (const float*)d_in[1];   // [2,128,1024]
    const float* c0  = (const float*)d_in[2];   // [2,128,1024]
    const float* Wih = (const float*)d_in[3];   // [2,4096,1024]
    const float* Whh = (const float*)d_in[4];   // [2,4096,1024]
    const float* bih = (const float*)d_in[5];   // [2,4096]
    const float* bhh = (const float*)d_in[6];   // [2,4096]

    // workspace layout
    char*   ws    = (char*)d_ws;
    __bf16* wih_b = (__bf16*)(ws);                          size_t o = (size_t)2 * WELEM * 2;
    __bf16* whh_b = (__bf16*)(ws + o);                      o += (size_t)2 * WELEM * 2;
    __bf16* xb    = (__bf16*)(ws + o);                      o += (size_t)T_STEPS * NH * 2;
    __bf16* hb    = (__bf16*)(ws + o);                      o += (size_t)2 * 2 * NH * 2;   // [L][slot][NH]
    float*  bias  = (float*) (ws + o);                      o += (size_t)2 * G4 * 4;
    float*  hf    = (float*) (ws + o);                      o += (size_t)2 * NH * 4;       // [L][NH]
    float*  cst   = (float*) (ws + o);                      // [L][NH]

    // one-time prep (fully parallel)
    k_cvt_bf16<<<2048, 256, 0, stream>>>(Wih, wih_b, (2 * WELEM) / 4);
    k_cvt_bf16<<<2048, 256, 0, stream>>>(Whh, whh_b, (2 * WELEM) / 4);
    k_cvt_bf16<<<8192, 256, 0, stream>>>(x,   xb,    (T_STEPS * NH) / 4);
    k_cvt_bf16<<<128,  256, 0, stream>>>(h0,      hb + 0 * 2 * NH, NH / 4);   // l0 slot0
    k_cvt_bf16<<<128,  256, 0, stream>>>(h0 + NH, hb + 1 * 2 * NH, NH / 4);   // l1 slot0
    k_bias<<<(2 * G4 + 255) / 256, 256, 0, stream>>>(bih, bhh, bias, 2 * G4);
    (void)hipMemcpyAsync(cst, c0, (size_t)2 * NH * 4, hipMemcpyDeviceToDevice, stream);

    for (int t = 0; t < T_STEPS; ++t) {
        int rs = t & 1, wsl = 1 - rs;
        const __bf16* xin0 = xb + (size_t)t * NH;
        __bf16* h0r = hb + 0 * 2 * NH + rs  * NH;
        __bf16* h0w = hb + 0 * 2 * NH + wsl * NH;
        __bf16* h1r = hb + 1 * 2 * NH + rs  * NH;
        __bf16* h1w = hb + 1 * 2 * NH + wsl * NH;
        // layer 0
        k_lstm_cell<<<128, 128, 0, stream>>>(xin0, h0r, cst,
                                             wih_b, whh_b, bias, h0w, hf);
        // layer 1 (input = layer 0's fresh bf16 h)
        k_lstm_cell<<<128, 128, 0, stream>>>(h0w, h1r, cst + NH,
                                             wih_b + WELEM, whh_b + WELEM, bias + G4,
                                             h1w, hf + NH);
    }

    // final output: layer-1 f32 hidden state at t = T-1
    (void)hipMemcpyAsync(d_out, hf + NH, (size_t)NH * 4, hipMemcpyDeviceToDevice, stream);
}
